// MDTA_1_33234456937063
// MI455X (gfx1250) — compile-verified, matrix operand order checked by reading, not run on hardware
//
#include <hip/hip_runtime.h>
#include <hip/hip_bf16.h>

// ---------- vector typedefs (all trivial ext_vector types, union-safe) ----------
typedef _Float16 v16h __attribute__((ext_vector_type(16)));
typedef _Float16 v8h  __attribute__((ext_vector_type(8)));
typedef float    v8f  __attribute__((ext_vector_type(8)));
typedef float    v8ff __attribute__((ext_vector_type(8)));
typedef float    v4f  __attribute__((ext_vector_type(4)));
typedef float    v2f  __attribute__((ext_vector_type(2)));
typedef int      v4i_t __attribute__((vector_size(16)));   // matches builtin param

union HF { v16h v; v8h h[2]; };
union F8 { v8ff v; v4f q[2]; float f[8]; };

#define HW   16384   // 128*128
#define BATCH 4
#define C_IN 192
#define C3   576
#define HEADS 4
#define DH   48      // C_IN / HEADS

// ---------- gfx1250 async global->LDS support (guarded) ----------
#if __has_builtin(__builtin_amdgcn_global_load_async_to_lds_b128)
#define MDTA_ASYNC 1
static __device__ __forceinline__ void mdta_wait_async() {
#if __has_builtin(__builtin_amdgcn_s_wait_asynccnt)
  __builtin_amdgcn_s_wait_asynccnt(0);
#else
  asm volatile("s_wait_asynccnt 0x0" ::: "memory");
#endif
}
#endif

// =====================================================================
// Kernel 1: generic  Y[o,n] = sum_c W[o,c] * X[c,n]   (per batch slice)
// 256 threads = 8 waves; output tile 64(o) x 64(n); K-step 32; f16 WMMA.
// O,C compile-time so the K loop (6 steps, 12 WMMAs) fully unrolls.
// =====================================================================
template <int O, int C>
__global__ __launch_bounds__(256) void mdta_gemm_wn_f16(
    const float* __restrict__ W, const float* __restrict__ X,
    float* __restrict__ Y)
{
  const int nt0 = blockIdx.x * 64;
  const int ot0 = blockIdx.y * 64;
  const int b   = blockIdx.z;
  X += (size_t)b * C * HW;
  Y += (size_t)b * O * HW;

  __shared__ __align__(16) _Float16 As[64 * 32];   // [o][k]
  __shared__ __align__(16) _Float16 Bs[64 * 32];   // [n][k]  (transposed staging)

  const int tid  = threadIdx.x;
  const int wave = tid >> 5;
  const int lane = tid & 31;
  const int m    = lane & 15;
  const int hi   = lane >> 4;

  const int mt  = wave >> 1;        // 0..3  (o tile)
  const int ntA = (wave & 1) * 2;   // 0 or 2 (first of two n tiles)

  v8f acc0 = {};
  v8f acc1 = {};

  const int sr = tid >> 2;          // 0..63 A staging row
  const int sc = (tid & 3) * 8;     // A staging k base
  const int xk = tid >> 3;          // 0..31 B staging k row
  const int xn = (tid & 7) * 8;     // B staging n base

#pragma unroll
  for (int kk = 0; kk < C; kk += 32) {
    { // stage A (f32 -> f16)
      F8 f;
      const float* wp = W + (size_t)(ot0 + sr) * C + kk + sc;
      f.q[0] = *(const v4f*)(wp);
      f.q[1] = *(const v4f*)(wp + 4);
      *(v8h*)&As[sr * 32 + sc] = __builtin_convertvector(f.v, v8h);
    }
    { // stage B transposed: Bs[n][k] = X[kk+xk][nt0+xn+i]
      F8 f;
      const float* xp = X + (size_t)(kk + xk) * HW + nt0 + xn;
      f.q[0] = *(const v4f*)(xp);
      f.q[1] = *(const v4f*)(xp + 4);
#pragma unroll
      for (int i = 0; i < 8; ++i)
        Bs[(xn + i) * 32 + xk] = (_Float16)f.f[i];
    }
    __syncthreads();

    HF a, b0, b1;
    const int arow = mt * 16 + m;
    a.h[0]  = *(const v8h*)&As[arow * 32 + hi * 8];
    a.h[1]  = *(const v8h*)&As[arow * 32 + 16 + hi * 8];
    const int br0 = ntA * 16 + m;
    const int br1 = (ntA + 1) * 16 + m;
    b0.h[0] = *(const v8h*)&Bs[br0 * 32 + hi * 8];
    b0.h[1] = *(const v8h*)&Bs[br0 * 32 + 16 + hi * 8];
    b1.h[0] = *(const v8h*)&Bs[br1 * 32 + hi * 8];
    b1.h[1] = *(const v8h*)&Bs[br1 * 32 + 16 + hi * 8];

    acc0 = __builtin_amdgcn_wmma_f32_16x16x32_f16(false, a.v, false, b0.v,
                                                  (short)0, acc0, false, false);
    acc1 = __builtin_amdgcn_wmma_f32_16x16x32_f16(false, a.v, false, b1.v,
                                                  (short)0, acc1, false, false);
    __syncthreads();
  }

#pragma unroll
  for (int r = 0; r < 8; ++r) {
    const int o = ot0 + mt * 16 + r + hi * 8;
    Y[(size_t)o * HW + nt0 + ntA * 16 + m]       = acc0[r];
    Y[(size_t)o * HW + nt0 + (ntA + 1) * 16 + m] = acc1[r];
  }
}

// =====================================================================
// Kernel 2: depthwise 3x3 conv, pad 1, LDS-staged rows.
// Block = 256 threads -> 2 output rows of 128.  Halo rows y0-1..y0+2 are
// staged into LDS with gfx1250 async global->LDS copies (ASYNCcnt path),
// then the 9-point stencil reads LDS only. grid: (64, B*576)
// =====================================================================
__global__ __launch_bounds__(256) void mdta_dwconv3x3(
    const float* __restrict__ Yin, const float* __restrict__ Wd,
    float* __restrict__ Z)
{
  const int bc = blockIdx.y;            // b*576 + ch
  const int ch = bc % C3;
  const int y0 = blockIdx.x * 2;        // first of the two output rows
  const int tid = threadIdx.x;
  const float* src = Yin + (size_t)bc * HW;
  const float* w   = Wd + ch * 9;

  __shared__ __align__(16) float tile[4 * 128];  // rows y0-1 .. y0+2

#ifdef MDTA_ASYNC
  if (tid < 128) {
    const int r  = tid >> 5;            // LDS row 0..3 (32 x 16B chunks each)
    const int cc = (tid & 31) * 4;      // float offset of this 16B chunk
    int gy = y0 - 1 + r;
    gy = gy < 0 ? 0 : (gy > 127 ? 127 : gy);   // clamp; OOB rows zeroed below
    __builtin_amdgcn_global_load_async_to_lds_b128(
        (__attribute__((address_space(1))) v4i_t*)(src + gy * 128 + cc),
        (__attribute__((address_space(3))) v4i_t*)&tile[r * 128 + cc], 0, 0);
  }
  mdta_wait_async();
  __syncthreads();
#else
  { // fallback: staged through registers
    for (int i = tid; i < 4 * 128; i += 256) {
      const int r = i >> 7;
      int gy = y0 - 1 + r;
      gy = gy < 0 ? 0 : (gy > 127 ? 127 : gy);
      tile[i] = src[gy * 128 + (i & 127)];
    }
    __syncthreads();
  }
#endif
  // zero the halo rows that fell outside the image
  if (y0 == 0   && tid < 128) tile[tid] = 0.f;
  if (y0 == 126 && tid < 128) tile[3 * 128 + tid] = 0.f;
  __syncthreads();

  const int ty = tid >> 7;              // 0..1
  const int x  = tid & 127;
  float s = 0.f;
#pragma unroll
  for (int dy = 0; dy < 3; ++dy) {
    const float* rw = &tile[(ty + dy) * 128];
    const float* wr = w + dy * 3;
    if (x > 0)   s += rw[x - 1] * wr[0];
    s += rw[x] * wr[1];
    if (x < 127) s += rw[x + 1] * wr[2];
  }
  Z[(size_t)bc * HW + (y0 + ty) * 128 + x] = s;
}

// =====================================================================
// Kernel 3: L2 row norms for q (sec 0) and k (sec 1) of e and d.
// grid = 3072 blocks; rid = ((t*2+sec)*B + b)*192 + ch
// =====================================================================
__global__ __launch_bounds__(256) void mdta_rownorm(
    const float* __restrict__ Ze, const float* __restrict__ Zd,
    float* __restrict__ norms)
{
  const int rid = blockIdx.x;
  const int ch  = rid % C_IN;
  int r = rid / C_IN;
  const int b   = r % BATCH; r /= BATCH;
  const int sec = r % 2;
  const int t   = r / 2;
  const float* src = (t ? Zd : Ze) + ((size_t)b * C3 + sec * C_IN + ch) * HW;
  float s = 0.f;
  for (int i = threadIdx.x; i < HW; i += 256) { const float v = src[i]; s += v * v; }
  __shared__ float red[256];
  red[threadIdx.x] = s;
  __syncthreads();
  for (int st = 128; st > 0; st >>= 1) {
    if (threadIdx.x < st) red[threadIdx.x] += red[threadIdx.x + st];
    __syncthreads();
  }
  if (threadIdx.x == 0) norms[rid] = sqrtf(red[0]);
}

// =====================================================================
// Kernel 4: Gram matrices S = Q K^T (48x48, K=16384) for 4 combos.
// grid (16 = b*h, 4 = combo), 288 threads = 9 waves (one 16x16 tile each).
// Fragments loaded directly from global (both operands row-major in K).
// =====================================================================
__global__ __launch_bounds__(288) void mdta_gram48(
    const float* __restrict__ Ze, const float* __restrict__ Zd,
    float* __restrict__ S)
{
  const int bh    = blockIdx.x;
  const int combo = blockIdx.y;
  const int b = bh >> 2, h = bh & 3;
  const int qd = (combo == 1) || (combo == 3);
  const int kd = (combo == 1) || (combo == 2);
  const float* Zq = qd ? Zd : Ze;
  const float* Zk = kd ? Zd : Ze;

  const int wave = threadIdx.x >> 5;
  const int lane = threadIdx.x & 31;
  const int m  = lane & 15, hi = lane >> 4;
  const int mt = wave / 3, nt = wave % 3;

  const float* qrow = Zq + ((size_t)b * C3 + 0     + h * DH + mt * 16 + m) * HW;
  const float* krow = Zk + ((size_t)b * C3 + C_IN  + h * DH + nt * 16 + m) * HW;

  v8f acc = {};
  for (int kk = 0; kk < HW; kk += 32) {
    if ((kk & 255) == 0 && kk + 2048 < HW) {
      __builtin_prefetch(qrow + kk + 2048, 0, 0);  // global_prefetch_b8
      __builtin_prefetch(krow + kk + 2048, 0, 0);
    }
    HF a, bb;
    F8 f;
    const float* qp = qrow + kk + hi * 8;
    f.q[0] = *(const v4f*)qp;        f.q[1] = *(const v4f*)(qp + 4);
    a.h[0] = __builtin_convertvector(f.v, v8h);
    f.q[0] = *(const v4f*)(qp + 16); f.q[1] = *(const v4f*)(qp + 20);
    a.h[1] = __builtin_convertvector(f.v, v8h);
    const float* kp = krow + kk + hi * 8;
    f.q[0] = *(const v4f*)kp;        f.q[1] = *(const v4f*)(kp + 4);
    bb.h[0] = __builtin_convertvector(f.v, v8h);
    f.q[0] = *(const v4f*)(kp + 16); f.q[1] = *(const v4f*)(kp + 20);
    bb.h[1] = __builtin_convertvector(f.v, v8h);

    acc = __builtin_amdgcn_wmma_f32_16x16x32_f16(false, a.v, false, bb.v,
                                                 (short)0, acc, false, false);
  }

  float* Sout = S + (((size_t)combo * BATCH + b) * HEADS + h) * (DH * DH);
#pragma unroll
  for (int r = 0; r < 8; ++r) {
    const int d = mt * 16 + r + hi * 8;
    const int e = nt * 16 + m;
    Sout[d * DH + e] = acc[r];
  }
}

// =====================================================================
// Kernel 5: softmax over e with temperature / (||q||*||k||) scaling, in place.
// One thread per (combo,b,h,d) row (48 wide). 3072 rows total.
// =====================================================================
__global__ __launch_bounds__(256) void mdta_softmax48(
    float* __restrict__ S, const float* __restrict__ norms,
    const float* __restrict__ temp)
{
  const int rid = blockIdx.x * 256 + threadIdx.x;
  if (rid >= 4 * BATCH * HEADS * DH) return;
  const int combo = rid / (BATCH * HEADS * DH);
  int r = rid % (BATCH * HEADS * DH);
  const int b = r / (HEADS * DH); r %= (HEADS * DH);
  const int h = r / DH;
  const int d = r % DH;
  const int qt = (combo == 1) || (combo == 3);
  const int kt = (combo == 1) || (combo == 2);

  const float qn = fmaxf(norms[((qt * 2 + 0) * BATCH + b) * C_IN + h * DH + d], 1e-12f);
  const float tm = temp[h];
  const float* knp = norms + ((kt * 2 + 1) * BATCH + b) * C_IN + h * DH;
  float* row = S + (((size_t)combo * BATCH + b) * HEADS + h) * (DH * DH) + d * DH;

  const float rq = tm / qn;
  float mx = -3.0e38f;
  for (int e = 0; e < DH; ++e) {
    const float v = row[e] * rq / fmaxf(knp[e], 1e-12f);
    mx = fmaxf(mx, v);
  }
  float sum = 0.f;
  for (int e = 0; e < DH; ++e) {
    const float v = row[e] * rq / fmaxf(knp[e], 1e-12f);
    sum += __expf(v - mx);
  }
  const float inv = 1.f / sum;
  for (int e = 0; e < DH; ++e) {
    const float v = row[e] * rq / fmaxf(knp[e], 1e-12f);
    row[e] = __expf(v - mx) * inv;
  }
}

// =====================================================================
// Kernel 6: U[b, h*48+d, n] = sum_e P[d,e] * V[b, h*48+e, n]
// f32 WMMA 16x16x4 (exact, K=48 = 12*4). grid (HW/128, b*h), 96 thr = 3 waves.
// =====================================================================
__global__ __launch_bounds__(96) void mdta_attnv48(
    const float* __restrict__ P,   // (b,h,48,48) attn for this combo
    const float* __restrict__ Zv,  // Z tensor holding the v section
    float* __restrict__ U)         // (b, 192, HW)
{
  const int bh = blockIdx.y;
  const int b = bh >> 2, h = bh & 3;
  const int n0base = blockIdx.x * 128;

  __shared__ __align__(16) float Ps[DH * DH];
  const float* Pin = P + (size_t)bh * (DH * DH);
  for (int i = threadIdx.x; i < DH * DH; i += 96) Ps[i] = Pin[i];
  __syncthreads();

  const int wave = threadIdx.x >> 5;   // == mt (0..2)
  const int lane = threadIdx.x & 31;
  const int m = lane & 15, hi = lane >> 4;

  const float* vbase = Zv + ((size_t)b * C3 + 2 * C_IN + h * DH) * HW;
  float* uo = U + ((size_t)b * C_IN + h * DH + wave * 16) * HW;

#if __has_builtin(__builtin_amdgcn_wmma_f32_16x16x4_f32)
  v2f afr[12];
#pragma unroll
  for (int kc = 0; kc < 12; ++kc) {
    const int k0 = kc * 4 + hi * 2;
    afr[kc][0] = Ps[(wave * 16 + m) * DH + k0];
    afr[kc][1] = Ps[(wave * 16 + m) * DH + k0 + 1];
  }
#pragma unroll 1
  for (int nt = 0; nt < 8; ++nt) {
    const int n = n0base + nt * 16 + m;
    v8f acc = {};
#pragma unroll
    for (int kc = 0; kc < 12; ++kc) {
      const int k0 = kc * 4 + hi * 2;
      v2f bf;
      bf[0] = vbase[(size_t)k0 * HW + n];
      bf[1] = vbase[(size_t)(k0 + 1) * HW + n];
      acc = __builtin_amdgcn_wmma_f32_16x16x4_f32(false, afr[kc], false, bf,
                                                  (short)0, acc, false, false);
    }
#pragma unroll
    for (int r = 0; r < 8; ++r)
      uo[(size_t)(r + hi * 8) * HW + n] = acc[r];
  }
#else
  // VALU fallback (same lane->element mapping as WMMA C/D layout)
#pragma unroll 1
  for (int nt = 0; nt < 8; ++nt) {
    const int n = n0base + nt * 16 + m;
#pragma unroll
    for (int r = 0; r < 8; ++r) {
      const int d = wave * 16 + r + hi * 8;
      float s = 0.f;
      for (int k = 0; k < DH; ++k) s += Ps[d * DH + k] * vbase[(size_t)k * HW + n];
      uo[(size_t)(r + hi * 8) * HW + n] = s;
    }
  }
#endif
}

// =====================================================================
// Host-side launch
// =====================================================================
extern "C" void kernel_launch(void* const* d_in, const int* in_sizes, int n_in,
                              void* d_out, int out_size, void* d_ws, size_t ws_size,
                              hipStream_t stream) {
  const float* e     = (const float*)d_in[0];
  const float* d     = (const float*)d_in[1];
  const float* temp  = (const float*)d_in[2];
  const float* w_qkv = (const float*)d_in[3];
  const float* w_dw  = (const float*)d_in[4];
  const float* w_p[4] = { (const float*)d_in[5], (const float*)d_in[6],
                          (const float*)d_in[7], (const float*)d_in[8] };
  float* out = (float*)d_out;
  float* ws  = (float*)d_ws;

  // workspace layout (floats)
  const size_t szY = (size_t)BATCH * C3 * HW;     // 37,748,736
  const size_t szU = (size_t)BATCH * C_IN * HW;   // 12,582,912
  const size_t szS = (size_t)4 * BATCH * HEADS * DH * DH;
  const size_t szN = (size_t)4 * BATCH * C_IN;
  float* Y     = ws;
  float* Ze    = ws + szY;
  float* Zd    = ws + 2 * szY;
  float* U     = ws + 3 * szY;
  float* S     = U + szU;
  float* norms = S + szS;
  if (ws_size < (3 * szY + szU + szS + szN) * sizeof(float)) return;

  const dim3 blk256(256);
  const dim3 gQKV(HW / 64, C3 / 64, BATCH);    // (256, 9, 4)
  const dim3 gPRJ(HW / 64, C_IN / 64, BATCH);  // (256, 3, 4)
  const dim3 gDW(64, BATCH * C3);              // (64, 2304) : 2 rows/block

  // QKV + depthwise for e and d (Y reused as scratch)
  mdta_gemm_wn_f16<C3, C_IN><<<gQKV, blk256, 0, stream>>>(w_qkv, e, Y);
  mdta_dwconv3x3<<<gDW, blk256, 0, stream>>>(Y, w_dw, Ze);
  mdta_gemm_wn_f16<C3, C_IN><<<gQKV, blk256, 0, stream>>>(w_qkv, d, Y);
  mdta_dwconv3x3<<<gDW, blk256, 0, stream>>>(Y, w_dw, Zd);

  // norms of q/k rows, Gram matrices, fused scale+softmax
  mdta_rownorm  <<<4 * BATCH * C_IN, blk256, 0, stream>>>(Ze, Zd, norms);
  mdta_gram48   <<<dim3(BATCH * HEADS, 4), 288, 0, stream>>>(Ze, Zd, S);
  mdta_softmax48<<<12, blk256, 0, stream>>>(S, norms, temp);

  // attn x V then projection for each of the 4 outputs
  const float* vsrc[4] = { Ze, Zd, Zd, Ze };  // ee, dd, ed(q_e,k_d,v_d), de(q_d,k_e,v_e)
  for (int c = 0; c < 4; ++c) {
    mdta_attnv48<<<dim3(HW / 128, BATCH * HEADS), 96, 0, stream>>>(
        S + (size_t)c * BATCH * HEADS * DH * DH, vsrc[c], U);
    mdta_gemm_wn_f16<C_IN, C_IN><<<gPRJ, blk256, 0, stream>>>(
        w_p[c], U, out + (size_t)c * szU);
  }
}
